// Net_30683246362843
// MI455X (gfx1250) — matrix operand order repaired and checked by reading, not compiled, not run on hardware
//
#include <hip/hip_runtime.h>
#include <hip/hip_bf16.h>

#define NN 50000
#define NE 800000
#define NF 128
#define DIM 32
#define HEADS 16
#define HD 512      // HEADS*DIM
#define NG 64

typedef __attribute__((ext_vector_type(16))) __bf16 v16bf;
typedef __attribute__((ext_vector_type(8)))  __bf16 v8bf;
typedef __attribute__((ext_vector_type(8)))  float  v8f;

// ---------- helpers ----------
static __device__ __forceinline__ float lrelu(float x) { return x > 0.f ? x : 0.2f * x; }

// monotone float<->uint key for atomicMax on floats
static __device__ __forceinline__ unsigned fkey(float f) {
  unsigned u = __float_as_uint(f);
  return (u & 0x80000000u) ? ~u : (u | 0x80000000u);
}
static __device__ __forceinline__ float funkey(unsigned k) {
  return __uint_as_float((k & 0x80000000u) ? (k ^ 0x80000000u) : ~k);
}

// A fragment from row-major f32 (convert in-register), ISA 16-bit A layout:
// lanes 0-15: row=lane, K {k0+0..7, k0+16..23}; lanes 16-31: row=lane-16, K {+8..15, +24..31}
static __device__ __forceinline__ v16bf load_a_f32(const float* __restrict__ A, int lda,
                                                   int k0, int lane) {
  int r  = lane & 15;
  int kb = k0 + ((lane >> 4) << 3);
  const float* p = A + (size_t)r * lda + kb;
  v16bf a;
#pragma unroll
  for (int i = 0; i < 8; ++i) a[i] = (__bf16)p[i];
#pragma unroll
  for (int i = 0; i < 8; ++i) a[8 + i] = (__bf16)p[16 + i];
  return a;
}

// A fragment from row-major bf16: two aligned 16-byte vector loads, no converts.
static __device__ __forceinline__ v16bf load_a_bf(const __bf16* __restrict__ A, int lda,
                                                  int k0, int lane) {
  int r  = lane & 15;
  int kb = k0 + ((lane >> 4) << 3);
  const __bf16* p = A + (size_t)r * lda + kb;
  v8bf lo = *(const v8bf*)p;          // 16B aligned: lda%8==0, kb%8==0
  v8bf hi = *(const v8bf*)(p + 16);
  v16bf a;
#pragma unroll
  for (int i = 0; i < 8; ++i) { a[i] = lo[i]; a[8 + i] = hi[i]; }
  return a;
}

// B fragment: 32x16 from pre-transposed bf16 weights WT[col][K].
// Per-lane data is K {kb+0..7, kb+16..23} = two aligned 16-byte vector loads.
static __device__ __forceinline__ v16bf load_bT(const __bf16* __restrict__ WT, int K,
                                                int k0, int c0, int lane) {
  int c  = c0 + (lane & 15);
  int kb = k0 + ((lane >> 4) << 3);
  const __bf16* p = WT + (size_t)c * K + kb;
  v8bf lo = *(const v8bf*)p;          // 16B-aligned: K%8==0, kb%8==0
  v8bf hi = *(const v8bf*)(p + 16);
  v16bf b;
#pragma unroll
  for (int i = 0; i < 8; ++i) { b[i] = lo[i]; b[8 + i] = hi[i]; }
  return b;
}

// ---------- one-shot weight prep: bf16 + transpose (fused rel|root) ----------
__global__ void k_prep_wcat(const float* __restrict__ Wrel, const float* __restrict__ Wroot,
                            int K, __bf16* __restrict__ WT) {
  int id = blockIdx.x * blockDim.x + threadIdx.x;   // 64*K threads
  if (id >= 64 * K) return;
  int c = id / K, k = id % K;
  float v = (c < 32) ? Wrel[(size_t)k * 32 + c] : Wroot[(size_t)k * 32 + (c - 32)];
  WT[(size_t)c * K + k] = (__bf16)v;
}

__global__ void k_prep_wg(const float* __restrict__ Wg, __bf16* __restrict__ WT) {
  int id = blockIdx.x * blockDim.x + threadIdx.x;   // 512*32 threads
  if (id >= HD * DIM) return;
  int c = id / DIM, k = id % DIM;
  WT[(size_t)c * DIM + k] = (__bf16)Wg[(size_t)k * HD + c];
}

// ---------- GEMM 1: Y[N,64] = X[N,128] @ [W1_rel | W1_root] ----------
__global__ void __launch_bounds__(32)
k_gemm_conv1(const float* __restrict__ X, const __bf16* __restrict__ W1T,
             float* __restrict__ Y) {
  int lane = threadIdx.x;
  int row0 = blockIdx.x * 16;
  v8f acc[4];
#pragma unroll
  for (int t = 0; t < 4; ++t)
#pragma unroll
    for (int i = 0; i < 8; ++i) acc[t][i] = 0.f;

  const float* Abase = X + (size_t)row0 * NF;
#pragma unroll
  for (int k0 = 0; k0 < NF; k0 += 32) {
    v16bf a = load_a_f32(Abase, NF, k0, lane);
#pragma unroll
    for (int t = 0; t < 4; ++t) {
      v16bf b = load_bT(W1T, NF, k0, t * 16, lane);
      acc[t] = __builtin_amdgcn_wmma_f32_16x16x32_bf16(false, a, false, b,
                                                       (short)0, acc[t], false, false);
    }
  }
  int cl = lane & 15, rb = (lane < 16) ? 0 : 8;
#pragma unroll
  for (int t = 0; t < 4; ++t)
#pragma unroll
    for (int i = 0; i < 8; ++i)
      Y[(size_t)(row0 + rb + i) * 64 + t * 16 + cl] = acc[t][i];
}

// ---------- GEMM 2: HG[N,512](bf16) = H1[N,32](bf16) @ Wg[32,512] ----------
__global__ void __launch_bounds__(32)
k_gemm_gat(const __bf16* __restrict__ H1, const __bf16* __restrict__ WgT,
           __bf16* __restrict__ HG) {
  int lane = threadIdx.x;
  int row0 = blockIdx.x * 16;
  v16bf a = load_a_bf(H1 + (size_t)row0 * DIM, DIM, 0, lane);
  int cl = lane & 15, rb = (lane < 16) ? 0 : 8;
#pragma unroll 4
  for (int t = 0; t < 32; ++t) {
    v16bf b = load_bT(WgT, DIM, 0, t * 16, lane);
    v8f c;
#pragma unroll
    for (int i = 0; i < 8; ++i) c[i] = 0.f;
    c = __builtin_amdgcn_wmma_f32_16x16x32_bf16(false, a, false, b, (short)0, c, false, false);
#pragma unroll
    for (int i = 0; i < 8; ++i)
      HG[(size_t)(row0 + rb + i) * HD + t * 16 + cl] = (__bf16)c[i];
  }
}

// ---------- GEMM 3: Y5[N,64] = H2[N,512](bf16) @ [W5_rel | W5_root] ----------
__global__ void __launch_bounds__(32)
k_gemm_conv5(const __bf16* __restrict__ H2, const __bf16* __restrict__ W5T,
             float* __restrict__ Y) {
  int lane = threadIdx.x;
  int row0 = blockIdx.x * 16;
  v8f acc[4];
#pragma unroll
  for (int t = 0; t < 4; ++t)
#pragma unroll
    for (int i = 0; i < 8; ++i) acc[t][i] = 0.f;

  const __bf16* Abase = H2 + (size_t)row0 * HD;
  for (int k0 = 0; k0 < HD; k0 += 32) {
    v16bf a = load_a_bf(Abase, HD, k0, lane);
#pragma unroll
    for (int t = 0; t < 4; ++t) {
      v16bf b = load_bT(W5T, HD, k0, t * 16, lane);
      acc[t] = __builtin_amdgcn_wmma_f32_16x16x32_bf16(false, a, false, b,
                                                       (short)0, acc[t], false, false);
    }
  }
  int cl = lane & 15, rb = (lane < 16) ? 0 : 8;
#pragma unroll
  for (int t = 0; t < 4; ++t)
#pragma unroll
    for (int i = 0; i < 8; ++i)
      Y[(size_t)(row0 + rb + i) * 64 + t * 16 + cl] = acc[t][i];
}

// ---------- edge scatter (GraphConv): agg[dst,c] += Y[src*64 + c], c<32 ----------
__global__ void k_scatter32(const float* __restrict__ Y, const long long* __restrict__ ei,
                            float* __restrict__ agg) {
  int id = blockIdx.x * blockDim.x + threadIdx.x;
  if (id >= NE * 32) return;
  int e = id >> 5, c = id & 31;
  int s = (int)ei[e], d = (int)ei[NE + e];
  atomicAdd(&agg[(size_t)d * 32 + c], Y[(size_t)s * 64 + c]);
}

// ---------- h1 = relu(agg + b + Y_root), emitted as bf16 for the GAT GEMM ----------
__global__ void k_relu_graphconv(const float* __restrict__ agg, const float* __restrict__ Y,
                                 const float* __restrict__ b, __bf16* __restrict__ H) {
  int id = blockIdx.x * blockDim.x + threadIdx.x;
  if (id >= NN * 32) return;
  int n = id >> 5, c = id & 31;
  float v = agg[id] + b[c] + Y[(size_t)n * 64 + 32 + c];
  H[id] = (__bf16)(v > 0.f ? v : 0.f);
}

// ---------- attention coefficients + self-loop max init ----------
__global__ void k_att(const __bf16* __restrict__ HG, const float* __restrict__ att_s,
                      const float* __restrict__ att_d, float* __restrict__ a_src,
                      float* __restrict__ a_dst, unsigned* __restrict__ emk) {
  int id = blockIdx.x * blockDim.x + threadIdx.x;
  if (id >= NN * HEADS) return;
  int n = id >> 4, h = id & 15;
  const v16bf* p = (const v16bf*)(HG + (size_t)n * HD + h * DIM);  // 64B-aligned row chunk
  v16bf h0 = p[0], h1 = p[1];
  float as = 0.f, ad = 0.f;
#pragma unroll
  for (int c = 0; c < 16; ++c) {
    float hv = (float)h0[c];
    as += hv * att_s[h * DIM + c];
    ad += hv * att_d[h * DIM + c];
  }
#pragma unroll
  for (int c = 0; c < 16; ++c) {
    float hv = (float)h1[c];
    as += hv * att_s[h * DIM + 16 + c];
    ad += hv * att_d[h * DIM + 16 + c];
  }
  a_src[id] = as;
  a_dst[id] = ad;
  emk[id] = fkey(lrelu(as + ad));  // self-loop score seeds the segment max
}

// ---------- edge pass: segment max ----------
__global__ void k_edge_max(const long long* __restrict__ ei, const float* __restrict__ a_src,
                           const float* __restrict__ a_dst, unsigned* __restrict__ emk) {
  int id = blockIdx.x * blockDim.x + threadIdx.x;
  if (id >= NE * HEADS) return;
  int e = id >> 4, h = id & 15;
  int s = (int)ei[e], d = (int)ei[NE + e];
  float v = lrelu(a_src[s * HEADS + h] + a_dst[d * HEADS + h]);
  atomicMax(&emk[d * HEADS + h], fkey(v));
}

// ---------- decode max, seed denominator with self-loop exp ----------
__global__ void k_denom_init(const unsigned* __restrict__ emk, const float* __restrict__ a_src,
                             const float* __restrict__ a_dst, float* __restrict__ emaxf,
                             float* __restrict__ denom) {
  int id = blockIdx.x * blockDim.x + threadIdx.x;
  if (id >= NN * HEADS) return;
  float m = funkey(emk[id]);
  emaxf[id] = m;
  denom[id] = __expf(lrelu(a_src[id] + a_dst[id]) - m);
}

// ---------- edge pass: denominator ----------
__global__ void k_edge_denom(const long long* __restrict__ ei, const float* __restrict__ a_src,
                             const float* __restrict__ a_dst, const float* __restrict__ emaxf,
                             float* __restrict__ denom) {
  int id = blockIdx.x * blockDim.x + threadIdx.x;
  if (id >= NE * HEADS) return;
  int e = id >> 4, h = id & 15;
  int s = (int)ei[e], d = (int)ei[NE + e];
  float v = lrelu(a_src[s * HEADS + h] + a_dst[d * HEADS + h]);
  atomicAdd(&denom[d * HEADS + h], __expf(v - emaxf[d * HEADS + h]));
}

// ---------- self-loop contribution initializes GAT output ----------
__global__ void k_self_out(const __bf16* __restrict__ HG, const float* __restrict__ a_src,
                           const float* __restrict__ a_dst, const float* __restrict__ emaxf,
                           const float* __restrict__ denom, float* __restrict__ G) {
  int id = blockIdx.x * blockDim.x + threadIdx.x;
  if (id >= NN * HD) return;
  int n = id >> 9, ch = id & (HD - 1), h = ch >> 5;
  int nh = n * HEADS + h;
  float alpha = __expf(lrelu(a_src[nh] + a_dst[nh]) - emaxf[nh]) / (denom[nh] + 1e-16f);
  G[id] = alpha * (float)HG[id];
}

// ---------- edge pass: weighted feature scatter (one thread = edge*head, 32 ch) ----------
__global__ void k_edge_gat(const long long* __restrict__ ei, const __bf16* __restrict__ HG,
                           const float* __restrict__ a_src, const float* __restrict__ a_dst,
                           const float* __restrict__ emaxf, const float* __restrict__ denom,
                           float* __restrict__ G) {
  int id = blockIdx.x * blockDim.x + threadIdx.x;
  if (id >= NE * HEADS) return;
  int e = id >> 4, h = id & 15;
  int s = (int)ei[e], d = (int)ei[NE + e];
  int dh = d * HEADS + h;
  float v = lrelu(a_src[s * HEADS + h] + a_dst[dh]);
  float alpha = __expf(v - emaxf[dh]) / (denom[dh] + 1e-16f);
  const v16bf* hp = (const v16bf*)(HG + (size_t)s * HD + h * DIM);  // 64B-aligned
  v16bf h0 = hp[0], h1 = hp[1];
  float* gp = G + (size_t)d * HD + h * DIM;
#pragma unroll
  for (int c = 0; c < 16; ++c) atomicAdd(&gp[c], alpha * (float)h0[c]);
#pragma unroll
  for (int c = 0; c < 16; ++c) atomicAdd(&gp[16 + c], alpha * (float)h1[c]);
}

// ---------- h2 = relu(G + bg), emitted bf16 (into retired HG buffer) ----------
__global__ void k_relu_bias512(const float* __restrict__ G, const float* __restrict__ bg,
                               __bf16* __restrict__ H2) {
  int id = blockIdx.x * blockDim.x + threadIdx.x;
  if (id >= NN * HD) return;
  float v = G[id] + bg[id & (HD - 1)];
  H2[id] = (__bf16)(v > 0.f ? v : 0.f);
}

// ---------- h3 = relu(agg + b + y_root), pooled into g by batch ----------
__global__ void k_conv5_pool(const float* __restrict__ agg, const float* __restrict__ Y,
                             const float* __restrict__ b, const long long* __restrict__ batch,
                             float* __restrict__ g) {
  int id = blockIdx.x * blockDim.x + threadIdx.x;
  if (id >= NN * 32) return;
  int n = id >> 5, c = id & 31;
  float v = agg[id] + b[c] + Y[(size_t)n * 64 + 32 + c];
  v = v > 0.f ? v : 0.f;
  atomicAdd(&g[(int)batch[n] * 32 + c], v);
}

// ---------- final MLP head + sigmoid (tiny: 64 graphs) ----------
__global__ void k_head(const float* __restrict__ g, const float* __restrict__ W1,
                       const float* __restrict__ b1, const float* __restrict__ W2,
                       const float* __restrict__ b2, float* __restrict__ out) {
  int gi = threadIdx.x;
  if (gi >= NG) return;
  float gv[DIM];
#pragma unroll
  for (int k = 0; k < DIM; ++k) gv[k] = g[gi * DIM + k];
  float hv[DIM];
  for (int j = 0; j < DIM; ++j) {
    float s = b1[j];
#pragma unroll
    for (int k = 0; k < DIM; ++k) s += gv[k] * W1[k * DIM + j];
    hv[j] = s > 0.f ? s : 0.f;
  }
  for (int o = 0; o < 2; ++o) {
    float s = b2[o];
#pragma unroll
    for (int j = 0; j < DIM; ++j) s += hv[j] * W2[j * 2 + o];
    out[gi * 2 + o] = 1.f / (1.f + __expf(-s));
  }
}

extern "C" void kernel_launch(void* const* d_in, const int* in_sizes, int n_in,
                              void* d_out, int out_size, void* d_ws, size_t ws_size,
                              hipStream_t stream) {
  const float*     x       = (const float*)d_in[0];
  const long long* ei      = (const long long*)d_in[1];
  const long long* batch   = (const long long*)d_in[2];
  const float*     W1_rel  = (const float*)d_in[3];
  const float*     b1_rel  = (const float*)d_in[4];
  const float*     W1_root = (const float*)d_in[5];
  const float*     Wg      = (const float*)d_in[6];
  const float*     att_src = (const float*)d_in[7];
  const float*     att_dst = (const float*)d_in[8];
  const float*     bg      = (const float*)d_in[9];
  const float*     W5_rel  = (const float*)d_in[10];
  const float*     b5_rel  = (const float*)d_in[11];
  const float*     W5_root = (const float*)d_in[12];
  const float*     Wfc1    = (const float*)d_in[13];
  const float*     bfc1    = (const float*)d_in[14];
  const float*     Wfc2    = (const float*)d_in[15];
  const float*     bfc2    = (const float*)d_in[16];
  float* out = (float*)d_out;

  // workspace carve-out
  char* ws = (char*)d_ws;
  size_t off = 0;
  auto carve = [&](size_t bytes) {
    char* p = ws + off;
    off = (off + bytes + 255) & ~(size_t)255;
    return p;
  };
  float*    y1    = (float*)carve((size_t)NN * 64 * 4);   // conv1 GEMM out; reused for conv5
  float*    agg   = (float*)carve((size_t)NN * 32 * 4);   // reused for both GraphConvs
  __bf16*   h1    = (__bf16*)carve((size_t)NN * 32 * 2);
  __bf16*   hg    = (__bf16*)carve((size_t)NN * HD * 2);  // GAT features; reused as bf16 h2
  float*    a_src = (float*)carve((size_t)NN * HEADS * 4);
  float*    a_dst = (float*)carve((size_t)NN * HEADS * 4);
  unsigned* emk   = (unsigned*)carve((size_t)NN * HEADS * 4);
  float*    emaxf = (float*)carve((size_t)NN * HEADS * 4);
  float*    denom = (float*)carve((size_t)NN * HEADS * 4);
  float*    gat   = (float*)carve((size_t)NN * HD * 4);
  float*    gpool = (float*)carve((size_t)NG * DIM * 4);
  __bf16*   W1T   = (__bf16*)carve((size_t)64 * NF * 2);  // [64][128]
  __bf16*   WgT   = (__bf16*)carve((size_t)HD * DIM * 2); // [512][32]
  __bf16*   W5T   = (__bf16*)carve((size_t)64 * HD * 2);  // [64][512]
  __bf16*   h2    = hg;                                   // HG retired before h2 is produced
  (void)ws_size; (void)in_sizes; (void)n_in; (void)out_size;

  const int ROWT = NN / 16;  // 3125 row tiles
  auto cdiv = [](long long a, long long b) { return (unsigned)((a + b - 1) / b); };

  // ---- weight prep (tiny, one-shot) ----
  k_prep_wcat<<<cdiv(64 * NF, 256), 256, 0, stream>>>(W1_rel, W1_root, NF, W1T);
  k_prep_wg<<<cdiv(HD * DIM, 256), 256, 0, stream>>>(Wg, WgT);
  k_prep_wcat<<<cdiv(64 * HD, 256), 256, 0, stream>>>(W5_rel, W5_root, HD, W5T);

  // ---- GraphConv 1 ----
  k_gemm_conv1<<<ROWT, 32, 0, stream>>>(x, W1T, y1);
  hipMemsetAsync(agg, 0, (size_t)NN * 32 * 4, stream);
  hipMemsetAsync(gpool, 0, (size_t)NG * DIM * 4, stream);
  k_scatter32<<<cdiv((long long)NE * 32, 256), 256, 0, stream>>>(y1, ei, agg);
  k_relu_graphconv<<<cdiv((long long)NN * 32, 256), 256, 0, stream>>>(agg, y1, b1_rel, h1);

  // ---- GATConv ----
  k_gemm_gat<<<ROWT, 32, 0, stream>>>(h1, WgT, hg);
  k_att<<<cdiv((long long)NN * HEADS, 256), 256, 0, stream>>>(hg, att_src, att_dst, a_src, a_dst, emk);
  k_edge_max<<<cdiv((long long)NE * HEADS, 256), 256, 0, stream>>>(ei, a_src, a_dst, emk);
  k_denom_init<<<cdiv((long long)NN * HEADS, 256), 256, 0, stream>>>(emk, a_src, a_dst, emaxf, denom);
  k_edge_denom<<<cdiv((long long)NE * HEADS, 256), 256, 0, stream>>>(ei, a_src, a_dst, emaxf, denom);
  k_self_out<<<cdiv((long long)NN * HD, 256), 256, 0, stream>>>(hg, a_src, a_dst, emaxf, denom, gat);
  k_edge_gat<<<cdiv((long long)NE * HEADS, 256), 256, 0, stream>>>(ei, hg, a_src, a_dst, emaxf, denom, gat);
  k_relu_bias512<<<cdiv((long long)NN * HD, 256), 256, 0, stream>>>(gat, bg, h2);

  // ---- GraphConv 5 ----
  k_gemm_conv5<<<ROWT, 32, 0, stream>>>(h2, W5T, y1);
  hipMemsetAsync(agg, 0, (size_t)NN * 32 * 4, stream);
  k_scatter32<<<cdiv((long long)NE * 32, 256), 256, 0, stream>>>(y1, ei, agg);

  // ---- pool + MLP head ----
  k_conv5_pool<<<cdiv((long long)NN * 32, 256), 256, 0, stream>>>(agg, y1, b5_rel, batch, gpool);
  k_head<<<1, 64, 0, stream>>>(gpool, Wfc1, bfc1, Wfc2, bfc2, out);
}
